// VectorizedSparseAttention_38689065402507
// MI455X (gfx1250) — matrix operand order repaired and checked by reading, not compiled, not run on hardware
//
#include <hip/hip_runtime.h>
#include <cstdint>

// =====================================================================
// VectorizedSparseAttention for MI455X (gfx1250, wave32, WMMA bf16 + TDM)
//
// Reference semantics notes:
//  * softmax is over axis=3 == 's' (query position in block), so
//    attn[s,d] = exp(S[s,d]) / colsum[d];  out += E @ (V rows / colsum)
//  * masked blocks => uniform 1/512 weights => contribute mean_d V[k,d,:]
// All GEMMs use v_wmma_f32_16x16x32_bf16 (8x the K-depth of the fp32
// 16x16x4 WMMA shape -> the compute-optimal path on CDNA5).
// K tiles are staged LDS-side with the Tensor Data Mover
// (tensor_load_to_lds + s_wait_tensorcnt), issued by wave 0.
// =====================================================================

typedef __bf16 bf16_t;
typedef __attribute__((ext_vector_type(16))) __bf16 v16bf;
typedef __attribute__((ext_vector_type(8)))  __bf16 v8bf;
typedef __attribute__((ext_vector_type(8)))  float  v8f;
typedef __attribute__((ext_vector_type(4)))  unsigned int u32x4;
typedef __attribute__((ext_vector_type(8)))  int i32x8;
typedef __attribute__((ext_vector_type(4)))  int i32x4;

#define NB    2
#define SEQ   8192
#define CH    128
#define BLK   512
#define NBLK  16
#define WIN   2
#define NKREL 5
#define SCALE 0.08838834764831845f   // 1/sqrt(128)

static __device__ __forceinline__ v8f zero8() {
  v8f z;
#pragma unroll
  for (int i = 0; i < 8; ++i) z[i] = 0.0f;
  return z;
}

static __device__ __forceinline__ v8f wmma_bf16(v16bf a, v16bf b, v8f c) {
  // (neg_a, A, neg_b, B, c_mod, C, reuse_a, reuse_b)
  return __builtin_amdgcn_wmma_f32_16x16x32_bf16(false, a, false, b,
                                                 (short)0, c, false, false);
}

// ---------------------------------------------------------------------
// Tensor Data Mover: 2D bf16 tile (rows x row_elems) from global (row
// stride row_stride elems) into contiguous LDS at lds_off.
// D# layout per CDNA5 ISA ch.8: group0 = {count=1, lds_addr,
// global_addr[56:0], type=2}; group1 = {data_size=2B, tensor_dim0/1,
// tile_dim0/1, dim0_stride}; groups 2/3 zero (<=2D tensor).
// ---------------------------------------------------------------------
#if defined(__clang_major__) && (__clang_major__ >= 23)
#define TDM_LOAD(g0, g1, g2, g3, z8) \
  __builtin_amdgcn_tensor_load_to_lds((g0), (g1), (g2), (g3), (z8), 0)
#else
#define TDM_LOAD(g0, g1, g2, g3, z8) \
  __builtin_amdgcn_tensor_load_to_lds((g0), (g1), (g2), (g3), 0)
#endif

static __device__ __forceinline__ void tdm_load_2d_bf16(uint32_t lds_off,
    const void* gptr, int rows, int row_elems, int row_stride) {
  const uint64_t ga = (uint64_t)(uintptr_t)gptr;
  u32x4 g0;
  g0[0] = 1u;                                           // count=1 user D#
  g0[1] = lds_off;                                      // lds_addr (bytes)
  g0[2] = (uint32_t)ga;                                 // global_addr[31:0]
  g0[3] = (uint32_t)((ga >> 32) & 0x01FFFFFFu) | (2u << 30);  // addr[56:32]|type=2
  i32x8 g1;
  g1[0] = 0x00010000;                                   // data_size=1 => 2 bytes
  g1[1] = (int)((row_elems & 0xFFFF) << 16);            // tensor_dim0[15:0]
  g1[2] = (int)((((unsigned)row_elems >> 16) & 0xFFFFu) |
                ((unsigned)(rows & 0xFFFF) << 16));     // dim0 hi | tensor_dim1 lo
  g1[3] = (int)((((unsigned)rows >> 16) & 0xFFFFu) |
                ((unsigned)(row_elems & 0xFFFF) << 16));// dim1 hi | tile_dim0
  g1[4] = rows & 0xFFFF;                                // tile_dim1 (tile_dim2=0)
  g1[5] = row_stride;                                   // tensor_dim0_stride[31:0]
  g1[6] = 0;
  g1[7] = 0;
  i32x4 gz;
  gz[0] = 0; gz[1] = 0; gz[2] = 0; gz[3] = 0;
  i32x8 z8;
#pragma unroll
  for (int i = 0; i < 8; ++i) z8[i] = 0;
  TDM_LOAD(g0, g1, gz, gz, z8);
  (void)z8;
}

// ---- A fragment: 16x32 bf16, row-major src (rows m0..m0+15, cols k0..k0+31)
// A k order used here (contiguous; B loaders match): lane m=lane&15, half=lane>>4,
// element i -> K = i + 16*half  => two contiguous 8-elem
// chunks at k = 16*half and k = 16*half+8  (vectorized 128-bit loads)
static __device__ __forceinline__ v16bf load_a_bf16(const bf16_t* src, int ld, int lane) {
  const int m = lane & 15, half = lane >> 4;
  const bf16_t* row = src + m * ld + 16 * half;
  v8bf lo = *(const v8bf*)(row);
  v8bf hi = *(const v8bf*)(row + 8);
  v16bf a;
#pragma unroll
  for (int i = 0; i < 8; ++i) { a[i] = lo[i]; a[8 + i] = hi[i]; }
  return a;
}

static __device__ __forceinline__ v16bf load_a_f32(const float* src, int ld, int lane) {
  const int m = lane & 15, half = lane >> 4;
  const float* row = src + m * ld;
  v16bf a;
#pragma unroll
  for (int i = 0; i < 16; ++i) {
    int k = i + 16 * half;
    a[i] = (bf16_t)row[k];
  }
  return a;
}

// ---- B fragment: 32x16 (K x N) from row-major src; lane n=lane&15,
// per-lane K range split 16/16 across half-waves (contiguous k, same order as the A loaders)
static __device__ __forceinline__ v16bf load_b_bf16(const bf16_t* src, int ld, int lane) {
  const int n = lane & 15, half = lane >> 4;
  v16bf b;
#pragma unroll
  for (int i = 0; i < 16; ++i) b[i] = src[(i + 16 * half) * ld + n];
  return b;
}

static __device__ __forceinline__ v16bf load_b_f32(const float* src, int ld, int lane) {
  const int n = lane & 15, half = lane >> 4;
  v16bf b;
#pragma unroll
  for (int i = 0; i < 16; ++i) b[i] = (bf16_t)src[(i + 16 * half) * ld + n];
  return b;
}

// ---- B = transpose of 16 key rows: B[k,n] = Krows[n][k]
// per-lane contiguous 16 bf16 => two 128-bit (ds) loads
static __device__ __forceinline__ v16bf load_bT_bf16(const bf16_t* src, int ld, int lane) {
  const int n = lane & 15, half = lane >> 4;
  const bf16_t* row = src + n * ld + 16 * half;
  v8bf lo = *(const v8bf*)(row);
  v8bf hi = *(const v8bf*)(row + 8);
  v16bf b;
#pragma unroll
  for (int i = 0; i < 8; ++i) { b[i] = lo[i]; b[8 + i] = hi[i]; }
  return b;
}

// =====================================================================
// Kernel 1: fused projections.  z=0: q = query@Wq+bq (N=128)
//                              z=1: [k|v] = key_value@Wkv+bkv (N=256)
// =====================================================================
__global__ __launch_bounds__(128) void proj_kernel(
    const float* __restrict__ query, const float* __restrict__ key_value,
    const float* __restrict__ Wq, const float* __restrict__ bq,
    const float* __restrict__ Wkv, const float* __restrict__ bkv,
    bf16_t* __restrict__ qbf, bf16_t* __restrict__ kbf, bf16_t* __restrict__ vbf) {
  const int lane = threadIdx.x & 31, w = threadIdx.x >> 5;
  const int z = blockIdx.z, by = blockIdx.y;
  if (z == 0 && by >= 2) return;
  const float* X = z ? key_value : query;
  const float* W = z ? Wkv : Wq;
  const float* bias = z ? bkv : bq;
  const int ldW = z ? 256 : 128;
  const int m0 = blockIdx.x * 64 + w * 16;
  const int n0 = by * 64;

  v8f acc[4];
#pragma unroll
  for (int j = 0; j < 4; ++j) acc[j] = zero8();

#pragma unroll
  for (int kc = 0; kc < 4; ++kc) {
    v16bf afrag = load_a_f32(X + (size_t)m0 * CH + kc * 32, CH, lane);
#pragma unroll
    for (int j = 0; j < 4; ++j) {
      v16bf bfrag = load_b_f32(W + (size_t)(kc * 32) * ldW + n0 + j * 16, ldW, lane);
      acc[j] = wmma_bf16(afrag, bfrag, acc[j]);
    }
  }

  const int half = lane >> 4, nlane = lane & 15;
#pragma unroll
  for (int j = 0; j < 4; ++j) {
    int ncol = n0 + j * 16 + nlane;
    float bv = bias[ncol];
#pragma unroll
    for (int r = 0; r < 8; ++r) {
      int m = m0 + r + 8 * half;
      float val = acc[j][r] + bv;
      if (z == 0)          qbf[(size_t)m * CH + ncol] = (bf16_t)val;
      else if (ncol < CH)  kbf[(size_t)m * CH + ncol] = (bf16_t)val;
      else                 vbf[(size_t)m * CH + (ncol - CH)] = (bf16_t)val;
    }
  }
}

// =====================================================================
// Kernel 2: per-(b,kblock) column sums of V (for disallowed-uniform term)
// =====================================================================
__global__ __launch_bounds__(128) void vsum_kernel(const bf16_t* __restrict__ vbf,
                                                   float* __restrict__ vsum) {
  const int blk = blockIdx.x;            // b*NBLK + kb
  const int b = blk / NBLK, kb = blk % NBLK;
  const int c = threadIdx.x;
  const bf16_t* base = vbf + (size_t)(b * SEQ + kb * BLK) * CH + c;
  float s = 0.0f;
  for (int d = 0; d < BLK; ++d) s += (float)base[(size_t)d * CH];
  vsum[(size_t)blk * CH + c] = s;
}

// =====================================================================
// Kernel 3: rs[b,qb,krel,d] = 1 / sum_s exp(S[s,d]*scale)
// 8 waves; each wave owns 64 query rows with Q fragments resident in VGPRs.
// K d-tiles staged into LDS by the TDM (wave 0); column sums reduced with
// ds_add_f32 atomics.
// =====================================================================
__global__ __launch_bounds__(256) void colsum_kernel(const bf16_t* __restrict__ qbf,
                                                     const bf16_t* __restrict__ kbf,
                                                     float* __restrict__ rs) {
  __shared__ bf16_t kt[16 * CH];   // 4 KB: one 16x128 K tile
  __shared__ float colsum[BLK];    // 2 KB
  const int tid = threadIdx.x, lane = tid & 31, w = tid >> 5;
  const int b = blockIdx.x / NBLK, qb = blockIdx.x % NBLK;
  const int kb = qb - WIN + (int)blockIdx.y;
  if (kb < 0 || kb >= NBLK) return;

  for (int i = tid; i < BLK; i += 256) colsum[i] = 0.0f;

  // Q fragments resident: 4 s-tiles x 4 k-chunks = 128 VGPRs
  v16bf qf[4][4];
  const bf16_t* qbase = qbf + (size_t)(b * SEQ + qb * BLK + w * 64) * CH;
#pragma unroll
  for (int st = 0; st < 4; ++st)
#pragma unroll
    for (int kc = 0; kc < 4; ++kc)
      qf[st][kc] = load_a_bf16(qbase + st * 16 * CH + kc * 32, CH, lane);

  const bf16_t* kbase = kbf + (size_t)(b * SEQ + kb * BLK) * CH;
  const uint32_t kt_off = (uint32_t)(uintptr_t)kt;
  for (int dt = 0; dt < 32; ++dt) {
    __syncthreads();                       // previous tile fully consumed
    if (w == 0) {
      tdm_load_2d_bf16(kt_off, kbase + (size_t)dt * 16 * CH, 16, CH, CH);
      __builtin_amdgcn_s_wait_tensorcnt((short)0);
    }
    __syncthreads();                       // tile landed in LDS

    v16bf bfrag[4];
#pragma unroll
    for (int kc = 0; kc < 4; ++kc) bfrag[kc] = load_bT_bf16(kt + kc * 32, CH, lane);

    float partial = 0.0f;
#pragma unroll
    for (int st = 0; st < 4; ++st) {
      v8f s = zero8();
#pragma unroll
      for (int kc = 0; kc < 4; ++kc) s = wmma_bf16(qf[st][kc], bfrag[kc], s);
#pragma unroll
      for (int r = 0; r < 8; ++r) partial += __expf(s[r] * SCALE);
    }
    partial += __shfl_xor(partial, 16, 32);          // fold half-waves (same column)
    if (lane < 16) atomicAdd(&colsum[dt * 16 + lane], partial);  // ds_add_f32
  }
  __syncthreads();

  float* out = rs + (size_t)((b * NBLK + qb) * NKREL + (int)blockIdx.y) * BLK;
  for (int i = tid; i < BLK; i += 256) out[i] = 1.0f / colsum[i];
}

// =====================================================================
// Kernel 4: attention output.  One workgroup per (b,qb); grid.y splits the
// 128 output channels in halves to stay within the 256-VGPR/wave budget at
// 16 waves/WGP.  K tile via TDM; V tile manually staged (needs rs scaling);
// E tiles round-trip through LDS (C/D -> A layout).
// =====================================================================
__global__ __launch_bounds__(512) void attn_kernel(
    const bf16_t* __restrict__ qbf, const bf16_t* __restrict__ kbf,
    const bf16_t* __restrict__ vbf, const float* __restrict__ rs,
    const float* __restrict__ vsum, bf16_t* __restrict__ attn) {
  __shared__ bf16_t kt[32 * CH];          // 8 KB
  __shared__ bf16_t vt[32 * CH];          // 8 KB (rs-scaled V)
  __shared__ bf16_t ew[16][2][16 * 32];   // 32 KB: per-wave E scratch
  __shared__ float rsl[BLK];              // 2 KB
  __shared__ float dis[64];

  const int tid = threadIdx.x, lane = tid & 31, w = tid >> 5;
  const int half = lane >> 4, nlane = lane & 15;
  const int b = blockIdx.x / NBLK, qb = blockIdx.x % NBLK;
  const int cg = blockIdx.y;              // channel half: tiles cg*4 .. cg*4+3

  // Q fragments resident: 2 s-tiles x 4 k-chunks
  v16bf qf[2][4];
  const bf16_t* qbase = qbf + (size_t)(b * SEQ + qb * BLK + w * 32) * CH;
#pragma unroll
  for (int st = 0; st < 2; ++st)
#pragma unroll
    for (int kc = 0; kc < 4; ++kc)
      qf[st][kc] = load_a_bf16(qbase + st * 16 * CH + kc * 32, CH, lane);

  v8f acc[2][4];
#pragma unroll
  for (int st = 0; st < 2; ++st)
#pragma unroll
    for (int ct = 0; ct < 4; ++ct) acc[st][ct] = zero8();

  const uint32_t kt_off = (uint32_t)(uintptr_t)kt;
  for (int krel = 0; krel < NKREL; ++krel) {
    int kb = qb - WIN + krel;
    if (kb < 0 || kb >= NBLK) continue;          // uniform across block
    __syncthreads();
    rsl[tid] = rs[(size_t)((b * NBLK + qb) * NKREL + krel) * BLK + tid];
    __syncthreads();

    const bf16_t* kbase = kbf + (size_t)(b * SEQ + kb * BLK) * CH;
    const bf16_t* vbase = vbf + (size_t)(b * SEQ + kb * BLK) * CH;

    for (int dt2 = 0; dt2 < 16; ++dt2) {         // 32 key rows per step
      __syncthreads();                           // previous tiles consumed
      if (w == 0)                                // async K tile DMA
        tdm_load_2d_bf16(kt_off, kbase + (size_t)dt2 * 32 * CH, 32, CH, CH);
      {                                          // V tile: copy + rs scale
        bf16_t tmp[8];
        *(uint4*)tmp = *(const uint4*)&vbase[(size_t)dt2 * 32 * CH + tid * 8];
        float rscale = rsl[dt2 * 32 + (tid >> 4)];
#pragma unroll
        for (int j = 0; j < 8; ++j) vt[tid * 8 + j] = (bf16_t)((float)tmp[j] * rscale);
      }
      if (w == 0) __builtin_amdgcn_s_wait_tensorcnt((short)0);
      __syncthreads();

      // E = exp(scale * Q Kt) for two s-tiles, staged to per-wave LDS
#pragma unroll
      for (int st = 0; st < 2; ++st) {
#pragma unroll
        for (int h = 0; h < 2; ++h) {
          v8f s = zero8();
#pragma unroll
          for (int kc = 0; kc < 4; ++kc) {
            v16bf bfrag = load_bT_bf16(kt + h * 16 * CH + kc * 32, CH, lane);
            s = wmma_bf16(qf[st][kc], bfrag, s);
          }
          bf16_t* dst = &ew[w][st][0];
#pragma unroll
          for (int r = 0; r < 8; ++r) {
            int m = r + 8 * half;
            dst[m * 32 + h * 16 + nlane] = (bf16_t)__expf(s[r] * SCALE);
          }
        }
      }
      // wave-local LDS write->read ordering (same wave, DS ops in order)
      asm volatile("s_wait_dscnt 0" ::: "memory");
      v16bf ef0 = load_a_bf16(&ew[w][0][0], 32, lane);
      v16bf ef1 = load_a_bf16(&ew[w][1][0], 32, lane);

#pragma unroll
      for (int ct = 0; ct < 4; ++ct) {
        v16bf vfrag = load_b_bf16(vt + (cg * 4 + ct) * 16, CH, lane);
        acc[0][ct] = wmma_bf16(ef0, vfrag, acc[0][ct]);
        acc[1][ct] = wmma_bf16(ef1, vfrag, acc[1][ct]);
      }
    }
  }

  // uniform 1/512 contribution from masked blocks
  __syncthreads();
  if (tid < 64) {
    int c = cg * 64 + tid;
    float t = 0.0f;
    for (int kb = 0; kb < NBLK; ++kb) {
      int d = kb - qb;
      if (d < -WIN || d > WIN) t += vsum[(size_t)(b * NBLK + kb) * CH + c];
    }
    dis[tid] = t * (1.0f / BLK);
  }
  __syncthreads();

#pragma unroll
  for (int st = 0; st < 2; ++st)
#pragma unroll
    for (int ct = 0; ct < 4; ++ct) {
      float dv = dis[ct * 16 + nlane];
      int ncol = cg * 64 + ct * 16 + nlane;
#pragma unroll
      for (int r = 0; r < 8; ++r) {
        int row = b * SEQ + qb * BLK + w * 32 + st * 16 + r + 8 * half;
        attn[(size_t)row * CH + ncol] = (bf16_t)(acc[st][ct][r] + dv);
      }
    }
}

// =====================================================================
// Kernel 5: output projection  out = attn @ Wo + bo  (fp32 result)
// =====================================================================
__global__ __launch_bounds__(128) void outproj_kernel(const bf16_t* __restrict__ attn,
                                                      const float* __restrict__ Wo,
                                                      const float* __restrict__ bo,
                                                      float* __restrict__ out) {
  const int lane = threadIdx.x & 31, w = threadIdx.x >> 5;
  const int m0 = blockIdx.x * 64 + w * 16;
  const int n0 = blockIdx.y * 64;

  v8f acc[4];
#pragma unroll
  for (int j = 0; j < 4; ++j) acc[j] = zero8();

#pragma unroll
  for (int kc = 0; kc < 4; ++kc) {
    v16bf a = load_a_bf16(attn + (size_t)m0 * CH + kc * 32, CH, lane);
#pragma unroll
    for (int j = 0; j < 4; ++j) {
      v16bf bfrag = load_b_f32(Wo + (size_t)(kc * 32) * CH + n0 + j * 16, CH, lane);
      acc[j] = wmma_bf16(a, bfrag, acc[j]);
    }
  }

  const int half = lane >> 4, nlane = lane & 15;
#pragma unroll
  for (int j = 0; j < 4; ++j) {
    int ncol = n0 + j * 16 + nlane;
    float bv = bo[ncol];
#pragma unroll
    for (int r = 0; r < 8; ++r) {
      int m = m0 + r + 8 * half;
      out[(size_t)m * CH + ncol] = acc[j][r] + bv;
    }
  }
}

// =====================================================================
extern "C" void kernel_launch(void* const* d_in, const int* in_sizes, int n_in,
                              void* d_out, int out_size, void* d_ws, size_t ws_size,
                              hipStream_t stream) {
  (void)in_sizes; (void)n_in; (void)out_size; (void)ws_size;
  const float* query     = (const float*)d_in[0];
  const float* key_value = (const float*)d_in[1];
  const float* Wq  = (const float*)d_in[2];
  const float* bq  = (const float*)d_in[3];
  const float* Wkv = (const float*)d_in[4];
  const float* bkv = (const float*)d_in[5];
  const float* Wo  = (const float*)d_in[6];
  const float* bo  = (const float*)d_in[7];

  const size_t MT = (size_t)NB * SEQ * CH;   // 2M elements
  bf16_t* qbf  = (bf16_t*)d_ws;
  bf16_t* kbf  = qbf + MT;
  bf16_t* vbf  = kbf + MT;
  bf16_t* attn = vbf + MT;
  float*  rs   = (float*)(attn + MT);
  float*  vsum = rs + (size_t)NB * NBLK * NKREL * BLK;

  proj_kernel<<<dim3(256, 4, 2), 128, 0, stream>>>(query, key_value, Wq, bq, Wkv, bkv,
                                                   qbf, kbf, vbf);
  vsum_kernel<<<NB * NBLK, CH, 0, stream>>>(vbf, vsum);
  colsum_kernel<<<dim3(NB * NBLK, NKREL), 256, 0, stream>>>(qbf, kbf, rs);
  attn_kernel<<<dim3(NB * NBLK, 2), 512, 0, stream>>>(qbf, kbf, vbf, rs, vsum, attn);
  outproj_kernel<<<dim3(256, 2), 128, 0, stream>>>(attn, Wo, bo, (float*)d_out);
}
